// HHNeurons_40114994545229
// MI455X (gfx1250) — compile-verified, matrix operand order checked by reading, not run on hardware
//
#include <hip/hip_runtime.h>
#include <hip/hip_bf16.h>

typedef __attribute__((ext_vector_type(16))) _Float16 v16h;
typedef __attribute__((ext_vector_type(8)))  float    v8f;

#define NN      4096
#define NTYPE   6
#define NSTEPS  1000
#define DTIME   0.01f
#define ROWSEG  4              // row-split factor (gridDim.y)
#define SEGROWS (NN / ROWSEG)  // 1024 rows per block

// ---------------- kernel 1: zero the cols scratch (d_ws is poisoned) -------
__global__ void hh_zero_cols(float* __restrict__ cols, int n) {
  int i = blockIdx.x * blockDim.x + threadIdx.x;
  if (i < n) cols[i] = 0.0f;
}

// ---------------- kernel 2: per-type masked column sums via WMMA -----------
// cols[t][j] = sum_i W[i][j] * (types[i][j] == t+1)
// D = Ones(16x32,f16) x MaskedTile(32x16,f16) + C(f32): every row of D holds
// the 16 column sums of the 32x16 tile; C carries f32 accumulation across
// chunks. Column sums are permutation-invariant in K, so the exact K striping
// of the B fragment cannot change the result. All 6 B fragments are built
// before the 6 WMMAs are issued so the WAR (B rebuild) hazard NOPs vanish.
__global__ __launch_bounds__(256) void hh_colsum_wmma(
    const float* __restrict__ W, const int* __restrict__ types,
    float* __restrict__ cols) {
  __shared__ float lw[256][17];   // padded: conflict-free 2-row column reads
  __shared__ int   lt[256][17];
  const int tid  = threadIdx.x;
  const int lane = tid & 31;
  const int wave = tid >> 5;         // 8 waves per block
  const int j0   = blockIdx.x * 16;  // 16-column stripe per block
  const int row0 = blockIdx.y * SEGROWS;
  const int nn   = lane & 15;        // output column within stripe
  const int g    = lane >> 4;        // lane half-group

  v16h ones;
#pragma unroll
  for (int e = 0; e < 16; ++e) ones[e] = (_Float16)1.0f;

  v8f zero = {};
  v8f acc[NTYPE];
#pragma unroll
  for (int t = 0; t < NTYPE; ++t) acc[t] = zero;

  const int cload = tid & 15;
  const int rload = tid >> 4;

  for (int rowbase = row0; rowbase < row0 + SEGROWS; rowbase += 256) {
    __syncthreads();  // previous chunk's LDS reads complete
    // cooperative coalesced staging of a 256x16 tile (W + types),
    // prefetching the next chunk (global_prefetch_b8 path)
    const bool more = (rowbase + 256) < (row0 + SEGROWS);
#pragma unroll
    for (int i = 0; i < 16; ++i) {
      int r = rload + i * 16;
      size_t gidx = (size_t)(rowbase + r) * NN + (size_t)(j0 + cload);
      lw[r][cload] = W[gidx];
      lt[r][cload] = types[gidx];
      if (more) {
        __builtin_prefetch(&W[gidx + (size_t)256 * NN], 0, 1);
        __builtin_prefetch(&types[gidx + (size_t)256 * NN], 0, 1);
      }
    }
    __syncthreads();

    // each wave owns rows [wave*32, wave*32+32) of the staged tile
    float wv[16]; int tv[16];
#pragma unroll
    for (int e = 0; e < 16; ++e) {
      int r = wave * 32 + g * 16 + e;
      wv[e] = lw[r][nn];
      tv[e] = lt[r][nn];
    }
    // build ALL per-type masked B fragments first ...
    v16h b[NTYPE];
#pragma unroll
    for (int t = 0; t < NTYPE; ++t) {
#pragma unroll
      for (int e = 0; e < 16; ++e)
        b[t][e] = (_Float16)((tv[e] == t + 1) ? wv[e] : 0.0f);
    }
    // ... then issue the 6 WMMAs back-to-back (disjoint B/acc: no hazards)
#pragma unroll
    for (int t = 0; t < NTYPE; ++t)
      acc[t] = __builtin_amdgcn_wmma_f32_16x16x32_f16(
          false, ones, false, b[t], (short)0, acc[t], false, false);
  }

  // D row 0 (lanes 0..15, VGPR0) holds the 16 column sums.
  if (lane < 16) {
#pragma unroll
    for (int t = 0; t < NTYPE; ++t)
      atomicAdd(&cols[t * NN + j0 + lane], acc[t][0]);
  }
}

// ---------------- kernel 3: persistent single-WGP HH time loop -------------
__global__ __launch_bounds__(1024) void hh_step_loop(
    const float* __restrict__ I_ext, const float* __restrict__ V0,
    const float* __restrict__ m0,    const float* __restrict__ h0,
    const float* __restrict__ n0,    const float* __restrict__ s0,
    const float* __restrict__ cols,  float* __restrict__ Vhist) {
  const int tid  = threadIdx.x;
  const int lane = tid & 31;   // wave32
  const int wid  = tid >> 5;   // 32 waves

  __shared__ float red[32];
  __shared__ float bcast;

  const float E_SYN[NTYPE] = {0.f, -70.f, -90.f, 0.f, 0.f, 0.f};
  const float TAUv [NTYPE] = {2.f, 5.f, 10.f, 100.f, 50.f, 30.f};
  const float SIGNv[NTYPE] = {-1.f, 1.f, 1.f, -1.f, -1.f, -1.f};
  float decay[NTYPE];
#pragma unroll
  for (int t = 0; t < NTYPE; ++t) decay[t] = 1.0f - DTIME / TAUv[t];

  // each thread owns 4 neurons: j = tid + q*1024 (coalesced)
  float V[4], m[4], h[4], nv[4], Ie[4];
  float s[NTYPE][4], cl[NTYPE][4];
#pragma unroll
  for (int q = 0; q < 4; ++q) {
    int j = tid + q * 1024;
    V[q]  = V0[j];  m[q]  = m0[j];  h[q] = h0[j];
    nv[q] = n0[j];  Ie[q] = I_ext[j];
#pragma unroll
    for (int t = 0; t < NTYPE; ++t) {
      s[t][q]  = s0[t * NN + j];
      cl[t][q] = cols[t * NN + j];
    }
  }

  for (int step = 0; step < NSTEPS; ++step) {
    // decay gates + fused signed synaptic-current contribution (one scalar)
    float contrib = 0.0f;
#pragma unroll
    for (int q = 0; q < 4; ++q) {
#pragma unroll
      for (int t = 0; t < NTYPE; ++t) {
        s[t][q] *= decay[t];
        contrib += SIGNv[t] * (0.1f * s[t][q] * (V[q] - E_SYN[t])) * cl[t][q];
      }
    }
    // block-wide scalar reduction: shfl within wave, LDS across waves
    float x = contrib;
#pragma unroll
    for (int off = 16; off > 0; off >>= 1) x += __shfl_xor(x, off, 32);
    if (lane == 0) red[wid] = x;
    __syncthreads();
    if (tid < 32) {
      float y = red[tid];
#pragma unroll
      for (int off = 16; off > 0; off >>= 1) y += __shfl_xor(y, off, 32);
      if (tid == 0) bcast = y;
    }
    __syncthreads();
    const float Isum = bcast;

#pragma unroll
    for (int q = 0; q < 4; ++q) {
      float v = V[q];
      float Ipre = Ie[q] + Isum;
      float am = (2.5f - 0.1f * v) / (__expf(2.5f - 0.1f * v) - 1.0f);
      float bm = 4.0f   * __expf(-v * (1.0f / 18.0f));
      float ah = 0.07f  * __expf(-v * (1.0f / 20.0f));
      float bh = 1.0f / (__expf(3.0f - 0.1f * v) + 1.0f);
      float an = (0.1f - 0.01f * v) / (__expf(1.0f - 0.1f * v) - 1.0f);
      float bn = 0.125f * __expf(-v * (1.0f / 80.0f));
      m[q]  += DTIME * (am * (1.0f - m[q])  - bm * m[q]);
      h[q]  += DTIME * (ah * (1.0f - h[q])  - bh * h[q]);
      nv[q] += DTIME * (an * (1.0f - nv[q]) - bn * nv[q]);
      float m3 = m[q] * m[q] * m[q];
      float n2 = nv[q] * nv[q];
      float INa = 120.0f * m3 * h[q] * (v - 50.0f);
      float IK  = 36.0f * (n2 * n2) * (v + 77.0f);
      float IL  = 0.3f * (v + 54.387f);
      float Vn = v + DTIME * (Ipre - INa - IK - IL);
      Vn = (Vn == Vn) ? fminf(fmaxf(Vn, -100.0f), 100.0f) : 0.0f;  // clip + nan_to_num
      V[q] = Vn;
      Vhist[(size_t)step * NN + (size_t)(tid + q * 1024)] = Vn;
    }
    // no 3rd barrier needed: next step's red[] writes are gated by this
    // step's 2nd barrier; bcast rewrite is gated by next step's 1st barrier.
  }
}

// ---------------------------- launch ---------------------------------------
extern "C" void kernel_launch(void* const* d_in, const int* in_sizes, int n_in,
                              void* d_out, int out_size, void* d_ws, size_t ws_size,
                              hipStream_t stream) {
  const float* I_ext = (const float*)d_in[0];
  const float* W     = (const float*)d_in[1];
  const float* V0    = (const float*)d_in[2];
  const float* m0    = (const float*)d_in[3];
  const float* h0    = (const float*)d_in[4];
  const float* n0    = (const float*)d_in[5];
  const float* s0    = (const float*)d_in[6];
  const int*   types = (const int*)d_in[7];
  // d_in[8] = T (device scalar); num_steps = T/DT = 1000, fixed by reference.

  float* cols  = (float*)d_ws;     // 6*4096 floats = 96 KB scratch
  float* Vhist = (float*)d_out;    // [1000, 4096] f32

  hh_zero_cols<<<(NTYPE * NN + 255) / 256, 256, 0, stream>>>(cols, NTYPE * NN);
  dim3 grid(NN / 16, ROWSEG, 1);
  hh_colsum_wmma<<<grid, 256, 0, stream>>>(W, types, cols);
  hh_step_loop<<<1, 1024, 0, stream>>>(I_ext, V0, m0, h0, n0, s0, cols, Vhist);
}
